// SelfAttentionWithGate_48928267436197
// MI455X (gfx1250) — matrix operand order repaired and checked by reading, not compiled, not run on hardware
//
#include <hip/hip_runtime.h>

typedef __attribute__((ext_vector_type(16))) _Float16 v16h;
typedef __attribute__((ext_vector_type(8)))  _Float16 v8h;
typedef __attribute__((ext_vector_type(4)))  _Float16 v4h;
typedef __attribute__((ext_vector_type(8)))  float    v8f;
typedef __attribute__((ext_vector_type(4)))  float    v4f;
typedef __attribute__((ext_vector_type(4)))  unsigned int u32x4;
typedef __attribute__((ext_vector_type(8)))  int      i32x8;
typedef __attribute__((ext_vector_type(4)))  int      i32x4;

#define S_  128
#define Q_  256
#define C_  256
#define H_  8
#define CH_ 32
#define E_  256   // H_*CH_

#if defined(__has_builtin)
#if __has_builtin(__builtin_amdgcn_tensor_load_to_lds)
#define USE_TDM 1
#endif
#endif
#ifndef USE_TDM
#define USE_TDM 0
#endif

static __device__ __forceinline__ v8f wmma16(v16h a, v16h b, v8f c) {
  // D = A(16x32 f16) * B(32x16 f16) + C(16x16 f32)
  return __builtin_amdgcn_wmma_f32_16x16x32_f16(false, a, false, b, (short)0, c, false, false);
}

// Cooperative fp32 weight [256x256] -> LDS f16 (row-major, = B^T operand layout).
static __device__ __forceinline__ void load_weight_lds(const float* __restrict__ W,
                                                       _Float16* lw, int tid) {
  const v4f* src = (const v4f*)W;
  v4h* dst = (v4h*)lw;
  #pragma unroll 4
  for (int i = tid; i < (E_ * C_) / 4; i += 256) {
    v4f f = src[i];
    v4h h;
    h.x = (_Float16)f.x; h.y = (_Float16)f.y;
    h.z = (_Float16)f.z; h.w = (_Float16)f.w;
    dst[i] = h;
  }
}

#if USE_TDM
// TDM: 1-D contiguous tile global -> LDS (nbytes multiple of 4, /4 must fit 16 bits).
// 6-arg builtin form: (g0, g1, g2, g3, g4, cpol)
static __device__ __forceinline__ void tdm_load_1d(const void* g, void* l, unsigned nbytes) {
  const unsigned long long ga = (unsigned long long)(uintptr_t)g;
  const unsigned la = (unsigned)(uintptr_t)l;       // low 32 bits = LDS byte offset
  const unsigned n4 = nbytes >> 2;                  // 4-byte elements
  u32x4 g0 = { 1u,                                  // count = 1 (valid descriptor)
               la,                                  // lds_addr
               (unsigned)ga,                        // global_addr[31:0]
               (unsigned)((ga >> 32) & 0x1ffffffu) | (2u << 30) }; // ga[56:32] | type=2
  i32x8 g1 = { (int)(2u << 16),                     // wg_mask=0, data_size=4B
               (int)((n4 & 0xffffu) << 16),         // tensor_dim0[15:0]
               (int)(((n4 >> 16) & 0xffffu) | (1u << 16)), // tensor_dim0[31:16], tensor_dim1=1
               (int)((n4 & 0xffffu) << 16),         // tile_dim0 = n4
               1,                                   // tile_dim1 = 1
               (int)n4,                             // tensor_dim0_stride
               0, 0 };
  i32x4 z4 = { 0, 0, 0, 0 };
  i32x8 z8 = { 0, 0, 0, 0, 0, 0, 0, 0 };
  __builtin_amdgcn_tensor_load_to_lds(g0, g1, z4, z4, z8, 0);
}
#endif

// ---------------------------------------------------------------------------
// Kernel 1: fused Q/K/V/G projections.  grid = (rows/128, 4), block = 256.
//   wid 0: qbuf[s][h][q][ch] f16, pre-scaled by 1/sqrt(CH)
//   wid 1: kbuf[s][h][q][ch] f16
//   wid 2: vT  [s][h][ch][q] f16   (transposed store, contiguous per lane)
//   wid 3: gateT[e][s*Q+q]   f16   sigmoid(x.wg + bg), transposed store
// ---------------------------------------------------------------------------
__global__ void k_proj(const float* __restrict__ X,
                       const float* __restrict__ wq, const float* __restrict__ wk,
                       const float* __restrict__ wv, const float* __restrict__ wg,
                       const float* __restrict__ bg,
                       _Float16* __restrict__ qbuf, _Float16* __restrict__ kbuf,
                       _Float16* __restrict__ vT,   _Float16* __restrict__ gateT) {
  extern __shared__ _Float16 lw[];            // [E_][C_] f16, 128 KB
  const int tid = threadIdx.x;
  const int wid = blockIdx.y;
  const float* W = (wid == 0) ? wq : (wid == 1) ? wk : (wid == 2) ? wv : wg;
  load_weight_lds(W, lw, tid);
  __syncthreads();

  const int lane = tid & 31, w = tid >> 5;
  const int half = lane >> 4, mrow = lane & 15;
  const int r0 = blockIdx.x * 128 + w * 16;   // 16 flattened (s,q) rows per wave

  // A operand: 16 rows x 256 cols of X, converted to f16, 8 K-steps of 32.
  const float* xr = X + (size_t)(r0 + mrow) * C_;
  v16h a[8];
  #pragma unroll
  for (int ks = 0; ks < 8; ++ks) {
    const int c0 = ks * 32 + 16 * half;
    v4f f0 = *(const v4f*)(xr + c0);
    v4f f1 = *(const v4f*)(xr + c0 + 4);
    v4f f2 = *(const v4f*)(xr + c0 + 8);
    v4f f3 = *(const v4f*)(xr + c0 + 12);
    v16h t;
    t[0]=(_Float16)f0.x; t[1]=(_Float16)f0.y; t[2]=(_Float16)f0.z; t[3]=(_Float16)f0.w;
    t[4]=(_Float16)f1.x; t[5]=(_Float16)f1.y; t[6]=(_Float16)f1.z; t[7]=(_Float16)f1.w;
    t[8]=(_Float16)f2.x; t[9]=(_Float16)f2.y; t[10]=(_Float16)f2.z; t[11]=(_Float16)f2.w;
    t[12]=(_Float16)f3.x; t[13]=(_Float16)f3.y; t[14]=(_Float16)f3.z; t[15]=(_Float16)f3.w;
    a[ks] = t;
  }

  const float scale = 0.17677669529663687f;   // 1/sqrt(32)

  #pragma unroll 1
  for (int nt = 0; nt < 16; ++nt) {
    const _Float16* bbase = lw + (size_t)(nt * 16 + mrow) * C_ + 16 * half;

    // Batch all B-operand LDS loads, then a dense WMMA chain.
    v16h b[8];
    #pragma unroll
    for (int ks = 0; ks < 8; ++ks)
      b[ks] = *(const v16h*)(bbase + ks * 32);

    v8f acc = {0.f,0.f,0.f,0.f,0.f,0.f,0.f,0.f};
    #pragma unroll
    for (int ks = 0; ks < 8; ++ks)
      acc = wmma16(a[ks], b[ks], acc);

    const int e  = nt * 16 + mrow;            // output column (E dim)
    const int hh = e >> 5, ch = e & 31;

    if (wid <= 1) {
      _Float16* dst = (wid == 0) ? qbuf : kbuf;
      const float sc = (wid == 0) ? scale : 1.0f;
      #pragma unroll
      for (int r = 0; r < 8; ++r) {
        const int Rg = r0 + 8 * half + r;
        const int s = Rg >> 8, qq = Rg & 255;
        dst[((size_t)(s * H_ + hh) * Q_ + qq) * CH_ + ch] = (_Float16)(acc[r] * sc);
      }
    } else if (wid == 2) {
      const int Rg0 = r0 + 8 * half;
      const int s = Rg0 >> 8, qq0 = Rg0 & 255;
      v8h hv;
      #pragma unroll
      for (int r = 0; r < 8; ++r) hv[r] = (_Float16)acc[r];
      *(v8h*)(vT + ((size_t)(s * H_ + hh) * CH_ + ch) * Q_ + qq0) = hv;
    } else {
      const float bge = bg[e];
      const int Rg0 = r0 + 8 * half;
      v8h hv;
      #pragma unroll
      for (int r = 0; r < 8; ++r) {
        const float t = __expf(-(acc[r] + bge));
        hv[r] = (_Float16)__builtin_amdgcn_rcpf(1.0f + t);
      }
      *(v8h*)(gateT + (size_t)e * (S_ * Q_) + Rg0) = hv;
    }
  }
}

// ---------------------------------------------------------------------------
// Kernel 2: attention per (s, h, 128-q block).  grid = (2, 8, 128), block = 256.
// logits^T = K_tile x Q^T so softmax dim is lane-local (+ one shfl_xor(16)).
// logits^T D-layout == A-operand layout of P for o = P x V (no reshuffle).
// K / V^T LDS fills go through the Tensor Data Mover when available.
// ---------------------------------------------------------------------------
__global__ void k_attn(const _Float16* __restrict__ qbuf, const _Float16* __restrict__ kbuf,
                       const _Float16* __restrict__ vT,   const _Float16* __restrict__ gateT,
                       const float* __restrict__ bias,    const float* __restrict__ mask,
                       _Float16* __restrict__ obuf) {
  __shared__ _Float16 lk[Q_ * CH_];           // K rows  [q][ch]  16 KB
  __shared__ _Float16 lv[CH_ * Q_];           // V^T     [ch][q]  16 KB
  const int s = blockIdx.z, h = blockIdx.y, qblk = blockIdx.x;
  const int tid = threadIdx.x;
  const int lane = tid & 31, w = tid >> 5;
  const size_t sh = (size_t)(s * H_ + h) * (Q_ * CH_);

#if USE_TDM
  if (w == 0) {                               // one wave issues both DMAs
    tdm_load_1d(kbuf + sh, lk, Q_ * CH_ * 2);
    tdm_load_1d(vT + sh, lv, Q_ * CH_ * 2);
    __builtin_amdgcn_s_wait_tensorcnt(0);
  }
#else
  {
    const uint4* gk = (const uint4*)(kbuf + sh);
    const uint4* gv = (const uint4*)(vT + sh);
    uint4* sk = (uint4*)lk;
    uint4* sv = (uint4*)lv;
    #pragma unroll 2
    for (int i = tid; i < (Q_ * CH_) / 8; i += 256) { sk[i] = gk[i]; sv[i] = gv[i]; }
  }
#endif
  __syncthreads();

  const int half = lane >> 4, mrow = lane & 15;
  const int qg = qblk * 128 + w * 16 + mrow;  // this lane's q row (column of logits^T)

  // B operand: Q^T, lane holds q-row qg, 32B contiguous.
  const v16h bq = *(const v16h*)(qbuf + sh + (size_t)qg * CH_ + 16 * half);

  const float* brow = bias + ((size_t)h * Q_ + qg) * Q_;
  __builtin_prefetch(brow, 0, 0);
  __builtin_prefetch(brow + 128, 0, 0);

  // logits^T: 16 M-tiles over k.
  v8f acc[16];
  #pragma unroll
  for (int t = 0; t < 16; ++t) {
    union { v16h v; v8h h2[2]; } ua;
    const _Float16* kr = lk + (size_t)(t * 16 + mrow) * CH_;
    ua.h2[0] = *(const v8h*)(kr + 16 * half);
    ua.h2[1] = *(const v8h*)(kr + 16 * half + 8);
    v8f z = {0.f,0.f,0.f,0.f,0.f,0.f,0.f,0.f};
    acc[t] = wmma16(ua.v, bq, z);
  }

  // + bias + (mask-1)*INF ; row max (lane-local then pair-combine).
  const float* mskr = mask + (size_t)s * Q_;
  float mx = -3.0e38f;
  #pragma unroll
  for (int t = 0; t < 16; ++t) {
    const int kb = t * 16 + 8 * half;
    v4f b0 = *(const v4f*)(brow + kb);
    v4f b1 = *(const v4f*)(brow + kb + 4);
    v4f m0 = *(const v4f*)(mskr + kb);
    v4f m1 = *(const v4f*)(mskr + kb + 4);
    float bb[8] = {b0.x,b0.y,b0.z,b0.w,b1.x,b1.y,b1.z,b1.w};
    float mm[8] = {m0.x,m0.y,m0.z,m0.w,m1.x,m1.y,m1.z,m1.w};
    #pragma unroll
    for (int r = 0; r < 8; ++r) {
      float l = acc[t][r] + bb[r] + (mm[r] - 1.0f) * 1.0e9f;
      acc[t][r] = l;
      mx = fmaxf(mx, l);
    }
  }
  mx = fmaxf(mx, __shfl_xor(mx, 16, 32));

  float sum = 0.0f;
  #pragma unroll
  for (int t = 0; t < 16; ++t)
    #pragma unroll
    for (int r = 0; r < 8; ++r) {
      float p = __expf(acc[t][r] - mx);
      acc[t][r] = p;                          // unnormalized prob
      sum += p;
    }
  sum += __shfl_xor(sum, 16, 32);
  const float inv = __builtin_amdgcn_rcpf(sum);   // applied post-GEMM

  // o = P x V : probs already in A-operand layout (tile 2t' -> elems 0..7, 2t'+1 -> 8..15).
  v8f o0 = {0.f,0.f,0.f,0.f,0.f,0.f,0.f,0.f};
  v8f o1 = {0.f,0.f,0.f,0.f,0.f,0.f,0.f,0.f};
  #pragma unroll
  for (int tp = 0; tp < 8; ++tp) {
    v16h ap;
    #pragma unroll
    for (int r = 0; r < 8; ++r) {
      ap[r]     = (_Float16)acc[2 * tp][r];
      ap[r + 8] = (_Float16)acc[2 * tp + 1][r];
    }
    const _Float16* v0 = lv + (size_t)mrow * Q_ + tp * 32 + 8 * half;
    const _Float16* v1 = lv + (size_t)(16 + mrow) * Q_ + tp * 32 + 8 * half;
    union { v16h v; v8h h2[2]; } ub0, ub1;
    ub0.h2[0] = *(const v8h*)v0; ub0.h2[1] = *(const v8h*)(v0 + 16);
    ub1.h2[0] = *(const v8h*)v1; ub1.h2[1] = *(const v8h*)(v1 + 16);
    o0 = wmma16(ap, ub0.v, o0);
    o1 = wmma16(ap, ub1.v, o1);
  }

  // normalize (shfl-broadcast inv per output row) + gate + store obuf[s][q][e] f16
  const int e0 = h * CH_ + mrow, e1 = e0 + 16;
  const int Rbase = s * Q_ + qblk * 128 + w * 16 + 8 * half;   // 8 consecutive rows
  const v8h gv0 = *(const v8h*)(gateT + (size_t)e0 * (S_ * Q_) + Rbase);
  const v8h gv1 = *(const v8h*)(gateT + (size_t)e1 * (S_ * Q_) + Rbase);
  #pragma unroll
  for (int r = 0; r < 8; ++r) {
    const float invR = __shfl(inv, 8 * half + r, 32);
    const size_t base = (size_t)(Rbase + r) * E_;
    obuf[base + e0] = (_Float16)(o0[r] * invR * (float)gv0[r]);
    obuf[base + e1] = (_Float16)(o1[r] * invR * (float)gv1[r]);
  }
}

// ---------------------------------------------------------------------------
// Kernel 3: out[q][s][c] = res + (o_gated . wo^T + bo), grid = 256, block = 256.
// ---------------------------------------------------------------------------
__global__ void k_out(const _Float16* __restrict__ obuf, const float* __restrict__ wo,
                      const float* __restrict__ bo, const float* __restrict__ res,
                      float* __restrict__ out) {
  extern __shared__ _Float16 lw[];            // wo [C_][E_] f16, 128 KB (= B^T layout)
  const int tid = threadIdx.x;
  load_weight_lds(wo, lw, tid);
  __syncthreads();

  const int lane = tid & 31, w = tid >> 5;
  const int half = lane >> 4, mrow = lane & 15;
  const int r0 = blockIdx.x * 128 + w * 16;

  const _Float16* orow = obuf + (size_t)(r0 + mrow) * E_;
  v16h a[8];
  #pragma unroll
  for (int ks = 0; ks < 8; ++ks) {
    union { v16h v; v8h h2[2]; } u;
    u.h2[0] = *(const v8h*)(orow + ks * 32 + 16 * half);
    u.h2[1] = *(const v8h*)(orow + ks * 32 + 16 * half + 8);
    a[ks] = u.v;
  }

  #pragma unroll 1
  for (int nt = 0; nt < 16; ++nt) {
    const _Float16* bbase = lw + (size_t)(nt * 16 + mrow) * E_ + 16 * half;

    v16h b[8];
    #pragma unroll
    for (int ks = 0; ks < 8; ++ks)
      b[ks] = *(const v16h*)(bbase + ks * 32);

    v8f acc = {0.f,0.f,0.f,0.f,0.f,0.f,0.f,0.f};
    #pragma unroll
    for (int ks = 0; ks < 8; ++ks)
      acc = wmma16(a[ks], b[ks], acc);

    const int c = nt * 16 + mrow;
    const float bc = bo[c];
    #pragma unroll
    for (int r = 0; r < 8; ++r) {
      const int Rg = r0 + 8 * half + r;
      const int s = Rg >> 8, q = Rg & 255;
      const size_t idx = ((size_t)q * S_ + s) * C_ + c;   // [Q][S][C]
      out[idx] = acc[r] + bc + res[idx];
    }
  }
}

// ---------------------------------------------------------------------------
extern "C" void kernel_launch(void* const* d_in, const int* in_sizes, int n_in,
                              void* d_out, int out_size, void* d_ws, size_t ws_size,
                              hipStream_t stream) {
  const float* X    = (const float*)d_in[0];   // [1,128,256,256]
  const float* mask = (const float*)d_in[1];   // [1,128,1,1,256]
  const float* bias = (const float*)d_in[2];   // [1,1,8,256,256]
  const float* res  = (const float*)d_in[3];   // [1,256,128,256]
  const float* wq   = (const float*)d_in[4];
  const float* wk   = (const float*)d_in[5];
  const float* wv   = (const float*)d_in[6];
  const float* wg   = (const float*)d_in[7];
  const float* bg   = (const float*)d_in[8];
  const float* wo   = (const float*)d_in[9];
  const float* bo   = (const float*)d_in[10];
  float* out = (float*)d_out;

  char* ws = (char*)d_ws;
  const size_t SLAB = (size_t)S_ * Q_ * E_ * sizeof(_Float16);  // 16.78 MB
  _Float16* qbuf  = (_Float16*)(ws);
  _Float16* kbuf  = (_Float16*)(ws + SLAB);
  _Float16* vT    = (_Float16*)(ws + 2 * SLAB);
  _Float16* gateT = (_Float16*)(ws + 3 * SLAB);
  _Float16* obuf  = (_Float16*)(ws + 4 * SLAB);

  const size_t lds = (size_t)E_ * C_ * sizeof(_Float16);        // 128 KB

  k_proj<<<dim3((S_ * Q_) / 128, 4, 1), 256, lds, stream>>>(
      X, wq, wk, wv, wg, bg, qbuf, kbuf, vT, gateT);
  k_attn<<<dim3(Q_ / 128, H_, S_), 256, 0, stream>>>(
      qbuf, kbuf, vT, gateT, bias, mask, obuf);
  k_out<<<dim3((S_ * Q_) / 128, 1, 1), 256, lds, stream>>>(
      obuf, wo, bo, res, out);
}
